// Encoder_90494960926886
// MI455X (gfx1250) — matrix operand order repaired and checked by reading, not compiled, not run on hardware
//
#include <hip/hip_runtime.h>

#define DD 128
#define BB 32

typedef __attribute__((ext_vector_type(16))) __bf16 v16bf;
typedef __attribute__((ext_vector_type(8)))  float  v8f;

// ---------------- token mean over rows: cls[b,d] = mean_t x[b,t,d] ----------------
__global__ void mean_rows_kernel(const float* __restrict__ x, float* __restrict__ cls, int N) {
    int b = blockIdx.x, d = threadIdx.x;
    const float* p = x + ((size_t)b * N) * DD + d;
    float s = 0.f;
    for (int t = 0; t < N; ++t) s += p[(size_t)t * DD];
    cls[b * DD + d] = s / (float)N;
}

// ---------------- Z = concat(clsA, clsB, X) @ W  via bf16 WMMA -------------------
// Block: 256 threads = 8 waves; 128-row tile x 128 cols. A/W staged to LDS as bf16.
// A (16x32 bf16) per-lane: lanes 0-15 -> row=lane, K kb..kb+7 & kb+16..kb+23,
// kb = kt*32 (+8 for lanes 16-31). B (32x16): col = lane&15, 16 K rows per lane, same k order as A.
// C/D f32: vgpr g -> row g (+8 for lanes 16-31), col = lane&15.
__global__ void __launch_bounds__(256)
gemm_prior_wmma(const float* __restrict__ X, const float* __restrict__ clsA,
                const float* __restrict__ clsB, const float* __restrict__ W,
                float* __restrict__ Z, int Nin, int Nout) {
    __shared__ __bf16 As[128 * 128];
    __shared__ __bf16 Ws[128 * 128];
    int b    = blockIdx.y;
    int tile = blockIdx.x * 128;
    int tid  = threadIdx.x;

    // stage W (f32 -> bf16), float4 global loads
    {
        const float4* Wv = (const float4*)W;
        for (int t = tid; t < 128 * 32; t += 256) {
            float4 w = Wv[t];
            __bf16* o = Ws + t * 4;
            o[0] = (__bf16)w.x; o[1] = (__bf16)w.y;
            o[2] = (__bf16)w.z; o[3] = (__bf16)w.w;
        }
    }
    // stage A tile, fusing the _token_prior concat row remap
    for (int t = tid; t < 128 * 32; t += 256) {
        int r = t >> 5, c4 = (t & 31) * 4;
        int g = tile + r;
        float4 v = make_float4(0.f, 0.f, 0.f, 0.f);
        if (g == 0)        v = *(const float4*)(clsA + b * DD + c4);
        else if (g == 1)   v = *(const float4*)(clsB + b * DD + c4);
        else if (g < Nout) v = *(const float4*)(X + ((size_t)b * Nin + (g - 2)) * DD + c4);
        __bf16* a = As + r * 128 + c4;
        a[0] = (__bf16)v.x; a[1] = (__bf16)v.y;
        a[2] = (__bf16)v.z; a[3] = (__bf16)v.w;
    }
    __syncthreads();

    int wv = tid >> 5, lane = tid & 31;
    int mrow = wv * 16 + (lane & 15);
    int koff = (lane >> 4) * 8;

    v16bf afr[4];
    #pragma unroll
    for (int kt = 0; kt < 4; ++kt) {
        int kb = kt * 32 + koff;
        #pragma unroll
        for (int i = 0; i < 8; ++i) {
            afr[kt][i]     = As[mrow * 128 + kb + i];
            afr[kt][i + 8] = As[mrow * 128 + kb + 16 + i];
        }
    }

    v8f acc[8];
    #pragma unroll
    for (int nt = 0; nt < 8; ++nt) acc[nt] = (v8f){};

    // kt outer, nt inner, fully unrolled: 8 independent accumulators per kt step
    // lets the scheduler overlap ds_load_b128 (next B frag) with current WMMA.
    #pragma unroll
    for (int kt = 0; kt < 4; ++kt) {
        int krow = kt * 32 + koff;
        #pragma unroll
        for (int nt = 0; nt < 8; ++nt) {
            v16bf bfr;
            #pragma unroll
            for (int i = 0; i < 16; ++i) bfr[i] = Ws[(krow + (i & 7) + ((i >> 3) << 4)) * 128 + nt * 16 + (lane & 15)];
            acc[nt] = __builtin_amdgcn_wmma_f32_16x16x32_bf16(
                false, afr[kt], false, bfr, (short)0, acc[nt], false, false);
        }
    }

    int col0  = lane & 15;
    int rbase = wv * 16 + (lane >> 4) * 8;
    if (tile + 128 <= Nout) {
        // full tile: branch-free stores (block-uniform condition)
        float* zrow = Z + ((size_t)b * Nout + tile + rbase) * DD + col0;
        #pragma unroll
        for (int nt = 0; nt < 8; ++nt) {
            #pragma unroll
            for (int g = 0; g < 8; ++g)
                zrow[(size_t)g * DD + nt * 16] = acc[nt][g];
        }
    } else {
        #pragma unroll
        for (int nt = 0; nt < 8; ++nt) {
            #pragma unroll
            for (int g = 0; g < 8; ++g) {
                int grow = tile + rbase + g;
                if (grow < Nout)
                    Z[((size_t)b * Nout + grow) * DD + nt * 16 + col0] = acc[nt][g];
            }
        }
    }
}

// ---------------- sim[b,t] = cls[b,:] . feat[b,t,:]  (wave-per-token) -------------
__global__ void sim_kernel(const float* __restrict__ cls, const float* __restrict__ feat,
                           float* __restrict__ sim, int N) {
    int b = blockIdx.y;
    int wv = threadIdx.x >> 5, lane = threadIdx.x & 31;
    int t  = blockIdx.x * 8 + wv;
    int tt = t < N ? t : N - 1;
    float4 f = *(const float4*)(feat + ((size_t)b * N + tt) * DD + lane * 4);
    float4 c = *(const float4*)(cls + b * DD + lane * 4);
    float p = f.x * c.x + f.y * c.y + f.z * c.z + f.w * c.w;
    #pragma unroll
    for (int off = 16; off > 0; off >>= 1) p += __shfl_xor(p, off, 32);
    if (lane == 0 && t < N) sim[(size_t)b * N + t] = p;
}

// ---------------- full descending argsort of sim (LDS bitonic, 4096 pad) ----------
// key = (ordered(score) << 32) | (~index): score desc, stable smallest-index first.
__global__ void __launch_bounds__(256)
topk_sort_kernel(const float* __restrict__ sim, int* __restrict__ idx, int N) {
    __shared__ unsigned long long keys[4096];
    int b = blockIdx.x, tid = threadIdx.x;
    for (int i = tid; i < 4096; i += 256) {
        unsigned long long kk = 0ull;
        if (i < N) {
            unsigned int u = __float_as_uint(sim[(size_t)b * N + i]);
            u = (u & 0x80000000u) ? ~u : (u | 0x80000000u);
            kk = ((unsigned long long)u << 32) |
                 (unsigned long long)(0xFFFFFFFFu - (unsigned)i);
        }
        keys[i] = kk;
    }
    __syncthreads();
    for (int k = 2; k <= 4096; k <<= 1) {
        for (int j = k >> 1; j > 0; j >>= 1) {
            for (int i = tid; i < 4096; i += 256) {
                int ij = i ^ j;
                if (ij > i) {
                    unsigned long long a = keys[i], c = keys[ij];
                    bool desc = ((i & k) == 0);
                    bool sw = desc ? (a < c) : (a > c);
                    if (sw) { keys[i] = c; keys[ij] = a; }
                }
            }
            __syncthreads();
        }
    }
    for (int i = tid; i < N; i += 256)
        idx[(size_t)b * N + i] =
            (int)(0xFFFFFFFFu - (unsigned)(keys[i] & 0xFFFFFFFFull));
}

// ---------------- fused = gather(top leftk of primary) ++ gather(top topk of sf) --
// one wave per row, float4 copies (128 floats = 32 lanes x float4)
__global__ void gather_rows_kernel(const float* __restrict__ zp, const float* __restrict__ zsf,
                                   const int* __restrict__ idxp, const int* __restrict__ idxsf,
                                   float* __restrict__ out, int ntoken, int leftk) {
    int b = blockIdx.y, j = blockIdx.x, lane = threadIdx.x;
    int src; const float* buf;
    if (j < leftk) { src = idxp[(size_t)b * ntoken + j];            buf = zp;  }
    else           { src = idxsf[(size_t)b * ntoken + (j - leftk)]; buf = zsf; }
    const float4* s = (const float4*)(buf + ((size_t)b * ntoken + src) * DD);
    float4*       d = (float4*)(out + ((size_t)b * ntoken + j) * DD);
    d[lane] = s[lane];
}

// ---------------- out = (a + b + c) / 3 (float4) ----------------------------------
__global__ void final_avg_kernel(const float4* __restrict__ a, const float4* __restrict__ b,
                                 const float4* __restrict__ c, float4* __restrict__ out, size_t n4) {
    size_t i = (size_t)blockIdx.x * blockDim.x + threadIdx.x;
    size_t stride = (size_t)gridDim.x * blockDim.x;
    for (; i < n4; i += stride) {
        float4 va = a[i], vb = b[i], vc = c[i];
        float4 r;
        r.x = (va.x + vb.x + vc.x) * (1.0f / 3.0f);
        r.y = (va.y + vb.y + vc.y) * (1.0f / 3.0f);
        r.z = (va.z + vb.z + vc.z) * (1.0f / 3.0f);
        r.w = (va.w + vb.w + vc.w) * (1.0f / 3.0f);
        out[i] = r;
    }
}

extern "C" void kernel_launch(void* const* d_in, const int* in_sizes, int n_in,
                              void* d_out, int out_size, void* d_ws, size_t ws_size,
                              hipStream_t stream) {
    (void)in_sizes; (void)n_in; (void)out_size; (void)ws_size;
    const float* x_s = (const float*)d_in[0];   // [32, 2048, 128]
    const float* x_f = (const float*)d_in[1];   // [32, 2048, 128]
    const float* W0  = (const float*)d_in[2];   // [2, 128, 128]
    const float* W1  = W0 + DD * DD;
    float* out = (float*)d_out;                 // [32, 2052, 128]

    const int L = 2048;
    const size_t CAP = (size_t)BB * 2052 * DD;
    float* ws = (float*)d_ws;
    float* P0 = ws + 0 * CAP;   // Z_s
    float* P1 = ws + 1 * CAP;   // Z_f
    float* P2 = ws + 2 * CAP;   // Z_sf (layer 0; persists as layer-1 sf input)
    float* P3 = ws + 3 * CAP;   // fused_s
    float* P4 = ws + 4 * CAP;   // fused_f
    float* P5 = ws + 5 * CAP;   // Z_sf (layer 1)
    float* cls = ws + 6 * CAP;  // 8 slots of [32,128]
    const size_t SIMCAP = (size_t)BB * 2052;
    float* sims = cls + 8 * BB * DD;            // 4 arrays of [32, ntoken]
    int*   idx  = (int*)(sims + 4 * SIMCAP);    // 4 arrays of [32, ntoken]

    float* c0 = cls + 0 * BB * DD;  // cls_s
    float* c1 = cls + 1 * BB * DD;  // cls_f
    float* c2 = cls + 2 * BB * DD;  // mean of projected s
    float* c3 = cls + 3 * BB * DD;  // mean of projected f
    float* c4 = cls + 4 * BB * DD;  // layer-1 cls_s
    float* c5 = cls + 5 * BB * DD;  // layer-1 cls_f
    float* c6 = cls + 6 * BB * DD;  // layer-1 cls_sf

    // ================= Layer 0 (Nin=2048 -> ntoken=2050, topk=205, leftk=1845) =====
    {
        const int Nin = L, Nout = L + 2, leftk = 1845;
        mean_rows_kernel<<<BB, DD, 0, stream>>>(x_s, c0, Nin);   // cls_s (== cls_sf, x_sf=x_s)
        mean_rows_kernel<<<BB, DD, 0, stream>>>(x_f, c1, Nin);   // cls_f
        dim3 ggrid((Nout + 127) / 128, BB);
        gemm_prior_wmma<<<ggrid, 256, 0, stream>>>(x_s, c1, c0, W0, P0, Nin, Nout); // s:  (cls_f, cls_sf, x_s)
        gemm_prior_wmma<<<ggrid, 256, 0, stream>>>(x_f, c0, c0, W0, P1, Nin, Nout); // f:  (cls_s, cls_sf, x_f)
        gemm_prior_wmma<<<ggrid, 256, 0, stream>>>(x_s, c0, c1, W0, P2, Nin, Nout); // sf: (cls_s, cls_f, x_sf)
        mean_rows_kernel<<<BB, DD, 0, stream>>>(P0, c2, Nout);
        mean_rows_kernel<<<BB, DD, 0, stream>>>(P1, c3, Nout);
        dim3 sgrid((Nout + 7) / 8, BB);
        sim_kernel<<<sgrid, 256, 0, stream>>>(c2, P0, sims + 0 * SIMCAP, Nout);
        sim_kernel<<<sgrid, 256, 0, stream>>>(c2, P2, sims + 1 * SIMCAP, Nout);
        sim_kernel<<<sgrid, 256, 0, stream>>>(c3, P1, sims + 2 * SIMCAP, Nout);
        sim_kernel<<<sgrid, 256, 0, stream>>>(c3, P2, sims + 3 * SIMCAP, Nout);
        for (int q = 0; q < 4; ++q)
            topk_sort_kernel<<<BB, 256, 0, stream>>>(sims + q * SIMCAP, idx + q * SIMCAP, Nout);
        dim3 tgrid(Nout, BB);
        gather_rows_kernel<<<tgrid, 32, 0, stream>>>(P0, P2, idx + 0 * SIMCAP, idx + 1 * SIMCAP, P3, Nout, leftk);
        gather_rows_kernel<<<tgrid, 32, 0, stream>>>(P1, P2, idx + 2 * SIMCAP, idx + 3 * SIMCAP, P4, Nout, leftk);
    }

    // ================= Layer 1 (Nin=2050 -> ntoken=2052, topk=205, leftk=1847) =====
    {
        const int Nin = L + 2, Nout = L + 4, leftk = 1847;
        mean_rows_kernel<<<BB, DD, 0, stream>>>(P3, c4, Nin);    // cls_s
        mean_rows_kernel<<<BB, DD, 0, stream>>>(P4, c5, Nin);    // cls_f
        mean_rows_kernel<<<BB, DD, 0, stream>>>(P2, c6, Nin);    // cls_sf
        dim3 ggrid((Nout + 127) / 128, BB);
        gemm_prior_wmma<<<ggrid, 256, 0, stream>>>(P3, c5, c6, W1, P0, Nin, Nout); // s
        gemm_prior_wmma<<<ggrid, 256, 0, stream>>>(P4, c4, c6, W1, P1, Nin, Nout); // f
        gemm_prior_wmma<<<ggrid, 256, 0, stream>>>(P2, c4, c5, W1, P5, Nin, Nout); // sf
        mean_rows_kernel<<<BB, DD, 0, stream>>>(P0, c2, Nout);
        mean_rows_kernel<<<BB, DD, 0, stream>>>(P1, c3, Nout);
        dim3 sgrid((Nout + 7) / 8, BB);
        sim_kernel<<<sgrid, 256, 0, stream>>>(c2, P0, sims + 0 * SIMCAP, Nout);
        sim_kernel<<<sgrid, 256, 0, stream>>>(c2, P5, sims + 1 * SIMCAP, Nout);
        sim_kernel<<<sgrid, 256, 0, stream>>>(c3, P1, sims + 2 * SIMCAP, Nout);
        sim_kernel<<<sgrid, 256, 0, stream>>>(c3, P5, sims + 3 * SIMCAP, Nout);
        for (int q = 0; q < 4; ++q)
            topk_sort_kernel<<<BB, 256, 0, stream>>>(sims + q * SIMCAP, idx + q * SIMCAP, Nout);
        dim3 tgrid(Nout, BB);
        gather_rows_kernel<<<tgrid, 32, 0, stream>>>(P0, P5, idx + 0 * SIMCAP, idx + 1 * SIMCAP, P3, Nout, leftk);
        gather_rows_kernel<<<tgrid, 32, 0, stream>>>(P1, P5, idx + 2 * SIMCAP, idx + 3 * SIMCAP, P4, Nout, leftk);

        size_t n4 = (size_t)BB * Nout * DD / 4;
        final_avg_kernel<<<1024, 256, 0, stream>>>((const float4*)P3, (const float4*)P4,
                                                   (const float4*)P5, (float4*)out, n4);
    }
}
